// TopologicalAttentionLayer_3229815407287
// MI455X (gfx1250) — matrix operand order repaired and checked by reading, not compiled, not run on hardware
//
#include <hip/hip_runtime.h>
#include <hip/hip_bf16.h>

#define BB   2
#define NN   2048
#define DD   256
#define HH   8
#define HDIM 32
#define EE   65536
#define KTOP 1024

typedef __attribute__((ext_vector_type(16))) _Float16 v16h;
typedef __attribute__((ext_vector_type(8)))  _Float16 v8h;
typedef __attribute__((ext_vector_type(4)))  _Float16 v4h;
typedef __attribute__((ext_vector_type(8)))  float    v8f;

// ---- WMMA fragment loaders (wave32, 16-bit, ISA 7.12.2) --------------------
// B fragment (32x16): lane=col, contiguous 16 halves starting at K = half*16.
__device__ __forceinline__ v16h ld_bfrag16(const _Float16* __restrict__ p) {
    v8h lo = *(const v8h*)p;
    v8h hi = *(const v8h*)(p + 8);
    return __builtin_shufflevector(lo, hi, 0, 1, 2, 3, 4, 5, 6, 7,
                                   8, 9, 10, 11, 12, 13, 14, 15);
}
// A fragment (16x32): lane=row, contiguous 16 halves:
//   K in [half*16, half*16+16)
__device__ __forceinline__ v16h ld_afrag16(const _Float16* __restrict__ row, int half) {
    v8h lo = *(const v8h*)(row + half * 16);
    v8h hi = *(const v8h*)(row + half * 16 + 8);
    return __builtin_shufflevector(lo, hi, 0, 1, 2, 3, 4, 5, 6, 7,
                                   8, 9, 10, 11, 12, 13, 14, 15);
}

// ---------------- f32 -> f16 conversion pass --------------------------------
__global__ void cvt_kernel(const float* __restrict__ src, _Float16* __restrict__ dst, int n4) {
    int i = blockIdx.x * blockDim.x + threadIdx.x;
    if (i < n4) {
        float4 v = ((const float4*)src)[i];
        v4h h;
        h[0] = (_Float16)v.x; h[1] = (_Float16)v.y;
        h[2] = (_Float16)v.z; h[3] = (_Float16)v.w;
        ((v4h*)dst)[i] = h;
    }
}

// ---------------- topo MLP: topo[b,n] = g2 . relu(g1 @ x + b1) + b2 ----------
__global__ void topo_kernel(const float* __restrict__ X, const float* __restrict__ g1w,
                            const float* __restrict__ g1b, const float* __restrict__ g2w,
                            const float* __restrict__ g2b, float* __restrict__ topo) {
    __shared__ float red[128];
    int node = blockIdx.x;               // 0 .. B*N-1
    int i = threadIdx.x;                 // hidden unit 0..127
    const float* xrow = X + (size_t)node * DD;
    const float* wrow = g1w + (size_t)i * DD;
    float acc = g1b[i];
    for (int k = 0; k < DD; ++k) acc = fmaf(xrow[k], wrow[k], acc);
    acc = fmaxf(acc, 0.0f) * g2w[i];
    red[i] = acc;
    __syncthreads();
    for (int s = 64; s > 0; s >>= 1) {
        if (i < s) red[i] += red[i + s];
        __syncthreads();
    }
    if (i == 0) topo[node] = red[0] + g2b[0];
}

// ---------------- top-k columns via rank ------------------------------------
__global__ void rank_kernel(const float* __restrict__ topo, float* __restrict__ colmask) {
    int idx = blockIdx.x * blockDim.x + threadIdx.x;
    if (idx >= BB * NN) return;
    int b = idx / NN, n = idx % NN;
    const float* t = topo + b * NN;
    float tv = t[n];
    int cnt = 0;
    for (int m = 0; m < NN; ++m) {
        float tm = t[m];
        cnt += (tm > tv) || (tm == tv && m < n);   // stable top-k tie-break
    }
    colmask[idx] = (cnt < KTOP) ? 1.0f : 0.0f;
}

// ---------------- edge mask -------------------------------------------------
__global__ void zero_kernel(float4* __restrict__ p, int n4) {
    int i = blockIdx.x * blockDim.x + threadIdx.x;
    if (i < n4) p[i] = make_float4(0.f, 0.f, 0.f, 0.f);
}
__global__ void scatter_kernel(const int* __restrict__ ei, float* __restrict__ edge) {
    int e = blockIdx.x * blockDim.x + threadIdx.x;
    if (e < EE) {
        int r = ei[e], c = ei[EE + e];
        edge[(size_t)r * NN + c] = 1.0f;
    }
}

// ---------------- sparse_mask output: broadcast over H (bandwidth kernel) ---
__global__ void mask_kernel(const float* __restrict__ edge, const float* __restrict__ colmask,
                            float* __restrict__ outMask) {
    size_t i = (size_t)blockIdx.x * blockDim.x + threadIdx.x;  // float4 index over [B,H,N,N/4]
    int m4 = (int)(i & (NN / 4 - 1));
    size_t j = i >> 9;                 // / (N/4)
    int n = (int)(j & (NN - 1));
    size_t k = j >> 11;                // / N
    int b = (int)(k >> 3);
    float4 ev = ((const float4*)(edge + (size_t)n * NN))[m4];
    float4 cv = ((const float4*)(colmask + (size_t)b * NN))[m4];
    float4 o;
    o.x = fmaxf(ev.x, cv.x); o.y = fmaxf(ev.y, cv.y);
    o.z = fmaxf(ev.z, cv.z); o.w = fmaxf(ev.w, cv.w);
    ((float4*)outMask)[i] = o;
}

// ---------------- unified WMMA GEMM: C = A @ W^T + bias ---------------------
// A: f16 [4096][256]; W: f16 [256][256] (row = output col). One wave computes
// a 16x64 slab (4 col tiles), A fragment amortized over 4 WMMAs per k-step.
// MODE 0: f16 out [node][d]   MODE 1: f16 out transposed [b][d][n]   MODE 2: f32 out
template <int MODE>
__global__ void gemm_kernel(const _Float16* __restrict__ A, const _Float16* __restrict__ W,
                            const float* __restrict__ bias, void* __restrict__ dstv) {
    int lane = threadIdx.x & 31;
    int wv   = threadIdx.x >> 5;
    int wid  = blockIdx.x * 8 + wv;          // 1024 waves = 256 rowTiles x 4 colGroups
    int rowTile = wid >> 2;
    int cgrp    = wid & 3;
    int half = lane >> 4;
    int l15  = lane & 15;

    const _Float16* arow = A + (size_t)(rowTile * 16 + l15) * DD;
    const _Float16* wrow[4];
#pragma unroll
    for (int c = 0; c < 4; ++c)
        wrow[c] = W + (size_t)(cgrp * 64 + c * 16 + l15) * DD + half * 16;

    v8f acc[4] = {};
    for (int ks = 0; ks < 8; ++ks) {
        int k0 = ks * 32;
        v16h a = ld_afrag16(arow + k0, half);
#pragma unroll
        for (int c = 0; c < 4; ++c) {
            v16h b = ld_bfrag16(wrow[c] + k0);
            acc[c] = __builtin_amdgcn_wmma_f32_16x16x32_f16(false, a, false, b, (short)0,
                                                            acc[c], false, false);
        }
    }
#pragma unroll
    for (int c = 0; c < 4; ++c) {
        int col = cgrp * 64 + c * 16 + l15;
        float bv = bias[col];
        if (MODE == 1) {                          // V^T: [b][d][n]; rows contiguous in n
            _Float16* dst = (_Float16*)dstv;
            int base = rowTile * 16;
            int bb = base >> 11, n0 = base & (NN - 1);
            v8h ph;
#pragma unroll
            for (int r = 0; r < 8; ++r) ph[r] = (_Float16)(acc[c][r] + bv);
            *(v8h*)(dst + ((size_t)bb * DD + col) * NN + n0 + half * 8) = ph;
        } else if (MODE == 0) {
            _Float16* dst = (_Float16*)dstv;
#pragma unroll
            for (int r = 0; r < 8; ++r) {
                int rg = rowTile * 16 + r + half * 8;
                dst[(size_t)rg * DD + col] = (_Float16)(acc[c][r] + bv);
            }
        } else {
            float* dst = (float*)dstv;
#pragma unroll
            for (int r = 0; r < 8; ++r) {
                int rg = rowTile * 16 + r + half * 8;
                dst[(size_t)rg * DD + col] = acc[c][r] + bv;
            }
        }
    }
}

// ---------------- flash-style masked attention ------------------------------
// one wave per (b, h, 16-row block); masked softmax (global softmax Z cancels)
__global__ void attn_kernel(const _Float16* __restrict__ qf, const _Float16* __restrict__ kf,
                            const _Float16* __restrict__ vt, const float* __restrict__ edge,
                            const float* __restrict__ colmask, _Float16* __restrict__ ao) {
    __shared__ alignas(16) _Float16 pst_all[8 * 16 * 32];  // wave-private P staging
    int lane = threadIdx.x & 31;
    int wv   = threadIdx.x >> 5;
    _Float16* pst = pst_all + wv * (16 * 32);
    int w   = blockIdx.x * 8 + wv;                   // 0 .. 2047
    int b   = w >> 10;                               // / (H * N/16)
    int rem = w & 1023;
    int h   = rem >> 7;
    int nt  = rem & 127;
    int n0  = nt * 16;
    int half = lane >> 4;
    int l15  = lane & 15;
    const float scale = 0.17677669529663687f;        // 1/sqrt(32)

    // Q A-fragment (loaded once): two b128 loads
    v16h aq = ld_afrag16(qf + (size_t)(b * NN + n0 + l15) * DD + h * HDIM, half);

    float mrun[8], lrun[8];
    v8f acc0 = {}, acc1 = {};
#pragma unroll
    for (int r = 0; r < 8; ++r) { mrun[r] = -3.0e38f; lrun[r] = 0.0f; }

    for (int mt = 0; mt < NN / 16; ++mt) {
        int m0 = mt * 16;
        // K B-fragment: col = key index, contiguous hd run -> two b128 loads
        v16h bk = ld_bfrag16(kf + (size_t)(b * NN + m0 + l15) * DD + h * HDIM + half * 16);
        v8f zc = {};
        v8f s = __builtin_amdgcn_wmma_f32_16x16x32_f16(false, aq, false, bk, (short)0, zc,
                                                       false, false);

        int col = m0 + l15;
        float cm = colmask[b * NN + col];
        float p[8];
#pragma unroll
        for (int r = 0; r < 8; ++r) {
            float sv = s[r] * scale;
            int row = n0 + r + half * 8;
            float mk = fmaxf(edge[(size_t)row * NN + col], cm);
            // row lives across 16 lanes of one half-wave -> butterfly within half
            float tm = sv;
            for (int off = 1; off < 16; off <<= 1) tm = fmaxf(tm, __shfl_xor(tm, off, 32));
            float nm  = fmaxf(mrun[r], tm);
            float fac = __expf(mrun[r] - nm);
            mrun[r] = nm;
            float pv = __expf(sv - nm) * mk;
            float ps = pv;
            for (int off = 1; off < 16; off <<= 1) ps += __shfl_xor(ps, off, 32);
            lrun[r] = lrun[r] * fac + ps;
            acc0[r] *= fac;
            acc1[r] *= fac;
            p[r] = pv;
        }
        // stage P tile (C layout -> LDS row-major 16x32, two m-tiles per PV step)
        int kpos = (mt & 1) * 16 + l15;
#pragma unroll
        for (int r = 0; r < 8; ++r)
            pst[(r + half * 8) * 32 + kpos] = (_Float16)p[r];

        if (mt & 1) {
            // A-fragment of P from LDS: two ds_load_b128
            v16h ap = ld_afrag16(pst + l15 * 32, half);
            int mb = m0 - 16;
            v16h bv0 = ld_bfrag16(vt + ((size_t)b * DD + h * HDIM + l15) * NN + mb + half * 16);
            v16h bv1 = ld_bfrag16(vt + ((size_t)b * DD + h * HDIM + 16 + l15) * NN + mb + half * 16);
            acc0 = __builtin_amdgcn_wmma_f32_16x16x32_f16(false, ap, false, bv0, (short)0, acc0,
                                                          false, false);
            acc1 = __builtin_amdgcn_wmma_f32_16x16x32_f16(false, ap, false, bv1, (short)0, acc1,
                                                          false, false);
        }
    }
#pragma unroll
    for (int r = 0; r < 8; ++r) {
        float inv = 1.0f / (lrun[r] + 1e-8f);
        int row = n0 + r + half * 8;
        _Float16* orow = ao + (size_t)(b * NN + row) * DD + h * HDIM;
        orow[l15]      = (_Float16)(acc0[r] * inv);
        orow[16 + l15] = (_Float16)(acc1[r] * inv);
    }
}

extern "C" void kernel_launch(void* const* d_in, const int* in_sizes, int n_in,
                              void* d_out, int out_size, void* d_ws, size_t ws_size,
                              hipStream_t stream) {
    const float* x    = (const float*)d_in[0];
    const int*   ei   = (const int*)d_in[1];
    const float* q_w  = (const float*)d_in[2];
    const float* q_b  = (const float*)d_in[3];
    const float* k_w  = (const float*)d_in[4];
    const float* k_b  = (const float*)d_in[5];
    const float* v_w  = (const float*)d_in[6];
    const float* v_b  = (const float*)d_in[7];
    const float* o_w  = (const float*)d_in[8];
    const float* o_b  = (const float*)d_in[9];
    const float* g1_w = (const float*)d_in[10];
    const float* g1_b = (const float*)d_in[11];
    const float* g2_w = (const float*)d_in[12];
    const float* g2_b = (const float*)d_in[13];

    char* ws = (char*)d_ws;
    size_t off = 0;
    auto alloc = [&](size_t bytes) {
        void* p = ws + off;
        off = (off + bytes + 255) & ~(size_t)255;
        return p;
    };
    float*    topo    = (float*)alloc((size_t)BB * NN * 4);
    float*    colmask = (float*)alloc((size_t)BB * NN * 4);
    float*    edge    = (float*)alloc((size_t)NN * NN * 4);
    _Float16* xh      = (_Float16*)alloc((size_t)BB * NN * DD * 2);
    _Float16* whq     = (_Float16*)alloc((size_t)DD * DD * 2);
    _Float16* whk     = (_Float16*)alloc((size_t)DD * DD * 2);
    _Float16* whv     = (_Float16*)alloc((size_t)DD * DD * 2);
    _Float16* who     = (_Float16*)alloc((size_t)DD * DD * 2);
    _Float16* qf      = (_Float16*)alloc((size_t)BB * NN * DD * 2);
    _Float16* kf      = (_Float16*)alloc((size_t)BB * NN * DD * 2);
    _Float16* vt      = (_Float16*)alloc((size_t)BB * DD * NN * 2);
    _Float16* ao      = (_Float16*)alloc((size_t)BB * NN * DD * 2);

    float* out     = (float*)d_out;
    float* outMask = out + (size_t)BB * NN * DD;

    // f32 -> f16 staging (once; removes conversions from GEMM hot loops)
    cvt_kernel<<<(BB * NN * DD / 4 + 255) / 256, 256, 0, stream>>>(x, xh, BB * NN * DD / 4);
    cvt_kernel<<<(DD * DD / 4 + 255) / 256, 256, 0, stream>>>(q_w, whq, DD * DD / 4);
    cvt_kernel<<<(DD * DD / 4 + 255) / 256, 256, 0, stream>>>(k_w, whk, DD * DD / 4);
    cvt_kernel<<<(DD * DD / 4 + 255) / 256, 256, 0, stream>>>(v_w, whv, DD * DD / 4);
    cvt_kernel<<<(DD * DD / 4 + 255) / 256, 256, 0, stream>>>(o_w, who, DD * DD / 4);

    topo_kernel<<<BB * NN, 128, 0, stream>>>(x, g1_w, g1_b, g2_w, g2_b, topo);
    rank_kernel<<<(BB * NN + 255) / 256, 256, 0, stream>>>(topo, colmask);
    zero_kernel<<<(NN * NN / 4 + 255) / 256, 256, 0, stream>>>((float4*)edge, NN * NN / 4);
    scatter_kernel<<<(EE + 255) / 256, 256, 0, stream>>>(ei, edge);
    mask_kernel<<<(unsigned)((size_t)BB * HH * NN * (NN / 4) / 256), 256, 0, stream>>>(
        edge, colmask, outMask);

    gemm_kernel<0><<<128, 256, 0, stream>>>(xh, whq, q_b, qf);
    gemm_kernel<0><<<128, 256, 0, stream>>>(xh, whk, k_b, kf);
    gemm_kernel<1><<<128, 256, 0, stream>>>(xh, whv, v_b, vt);
    attn_kernel<<<256, 256, 0, stream>>>(qf, kf, vt, edge, colmask, ao);
    gemm_kernel<2><<<128, 256, 0, stream>>>(ao, who, o_b, out);
}
